// PFNet6_15814069584534
// MI455X (gfx1250) — matrix operand order repaired and checked by reading, not compiled, not run on hardware
//
#include <hip/hip_runtime.h>
#include <hip/hip_bf16.h>

typedef __attribute__((ext_vector_type(16))) _Float16 v16h;
typedef __attribute__((ext_vector_type(8)))  _Float16 v8h;
typedef __attribute__((ext_vector_type(4)))  _Float16 v4h;
typedef __attribute__((ext_vector_type(8)))  float    v8f;

struct P47 { const void* p[47]; };

#define SELU_ALPHA 1.6732632423543772f
#define SELU_SCALE 1.0507009873554805f

// Branchless SELU: pos-term is 0 for v<=0, neg-term is 0 for v>=0 (exp(0)-1 == 0).
__device__ __forceinline__ float selu_f(float v) {
  float neg = SELU_SCALE * SELU_ALPHA * (__expf(fminf(v, 0.f)) - 1.f);
  float pos = SELU_SCALE * fmaxf(v, 0.f);
  return pos + neg;
}

__device__ __forceinline__ v8f wmma16(v16h a, v16h b, v8f c) {
  return __builtin_amdgcn_wmma_f32_16x16x32_f16(false, a, false, b, (short)0, c, false, false);
}

// A fragment from a row-major f16 LDS tile (row stride ldm halves, col offset koff).
// Lane holds row M = lane&15; element e holds K = 16*(lane>=16) + e,
// i.e. one contiguous 16-half run -> two ds_load_b128.
__device__ __forceinline__ v16h frag_a(const _Float16* act, int lane, int ldm, int koff) {
  int m = lane & 15;
  int kh = (lane >> 4) & 1;
  const _Float16* base = act + m * ldm + koff + (kh << 4);
  v8h lo = *(const v8h*)(base);
  v8h hi = *(const v8h*)(base + 8);
  return __builtin_shufflevector(lo, hi, 0, 1, 2, 3, 4, 5, 6, 7, 8, 9, 10, 11, 12, 13, 14, 15);
}

// B fragment from weights pre-packed in fragment order:
// packed[(((kt*ntN + nt)*2 + kh) << 8) + n0*16 + e]  (contiguous in e -> 2x ds_load_b128)
__device__ __forceinline__ v16h frag_bp(const _Float16* pk, int lane, int ntN, int kt, int nt) {
  int n0 = lane & 15, kh = (lane >> 4) & 1;
  const _Float16* base = pk + ((((kt * ntN + nt) << 1) + kh) << 8) + (n0 << 4);
  v8h lo = *(const v8h*)(base);
  v8h hi = *(const v8h*)(base + 8);
  return __builtin_shufflevector(lo, hi, 0, 1, 2, 3, 4, 5, 6, 7, 8, 9, 10, 11, 12, 13, 14, 15);
}

// Stage a K x Ncols f32 weight matrix (K multiple of 32, Ncols multiple of 16)
// into packed B-fragment order as f16.
__device__ __forceinline__ void stage_pack(_Float16* dst, const float* W, int K, int Ncols) {
  int ntN = Ncols >> 4;
  int total = K * Ncols;
  for (int idx = threadIdx.x; idx < total; idx += 256) {
    int e = idx & 15, n0 = (idx >> 4) & 15, kh = (idx >> 8) & 1;
    int rest = idx >> 9;
    int nt = rest % ntN, kt = rest / ntN;
    dst[idx] = (_Float16)W[(kt * 32 + kh * 16 + e) * Ncols + nt * 16 + n0];
  }
}

// ---------------------------------------------------------------------------
// Kernel 1: node input MLP (3->32, 7x(32->32), SELU between) fused with @conv1_W
// ---------------------------------------------------------------------------
__global__ __launch_bounds__(256) void k_inp_mlp(P47 P, float* __restrict__ xw1, int N) {
  __shared__ _Float16 sWp[8][1024];   // packed: [0..6]=inp layers 1..7, [7]=conv1_W
  __shared__ float sB[7][32];
  __shared__ float sW0[96];
  __shared__ float sB0[32];
  __shared__ _Float16 sAct[8][16 * 40];

  for (int l = 0; l < 7; ++l) {
    stage_pack(sWp[l], (const float*)P.p[5 + 2 * l], 32, 32);
    if (threadIdx.x < 32) sB[l][threadIdx.x] = ((const float*)P.p[6 + 2 * l])[threadIdx.x];
  }
  stage_pack(sWp[7], (const float*)P.p[19], 32, 32);  // conv1_W
  if (threadIdx.x < 96) sW0[threadIdx.x] = ((const float*)P.p[3])[threadIdx.x];
  if (threadIdx.x < 32) sB0[threadIdx.x] = ((const float*)P.p[4])[threadIdx.x];
  __syncthreads();

  int wave = threadIdx.x >> 5, lane = threadIdx.x & 31;
  int ntiles = (N + 15) >> 4;
  int tile = blockIdx.x * 8 + wave;
  if (tile >= ntiles) return;

  const float* x = (const float*)P.p[0];
  _Float16* act = sAct[wave];
  int nbase = tile << 4;

  // layer 0: 3 -> 32 + SELU (lane-scalar)
  for (int idx = lane; idx < 512; idx += 32) {
    int r = idx >> 5, c = idx & 31;
    int node = nbase + r; if (node > N - 1) node = N - 1;
    const float* xr = x + node * 3;
    float acc = sB0[c] + xr[0] * sW0[c] + xr[1] * sW0[32 + c] + xr[2] * sW0[64 + c];
    act[r * 40 + c] = (_Float16)selu_f(acc);
  }

  int n0 = lane & 15, kh = (lane >> 4) & 1;

  // layers 1..7 (WMMA); SELU on all but last
  for (int l = 0; l < 7; ++l) {
    v16h a = frag_a(act, lane, 40, 0);
    v8f c0 = {}, c1 = {};
    c0 = wmma16(a, frag_bp(sWp[l], lane, 2, 0, 0), c0);
    c1 = wmma16(a, frag_bp(sWp[l], lane, 2, 0, 1), c1);
    bool doact = (l < 6);
#pragma unroll
    for (int r = 0; r < 8; ++r) {
      int m = (kh << 3) + r;
      float v0 = c0[r] + sB[l][n0];
      float v1 = c1[r] + sB[l][n0 + 16];
      if (doact) { v0 = selu_f(v0); v1 = selu_f(v1); }
      act[m * 40 + n0]      = (_Float16)v0;
      act[m * 40 + n0 + 16] = (_Float16)v1;
    }
  }

  // xw1 = h0 @ conv1_W  (no bias, no act) -> f32 global
  {
    v16h a = frag_a(act, lane, 40, 0);
    v8f c0 = {}, c1 = {};
    c0 = wmma16(a, frag_bp(sWp[7], lane, 2, 0, 0), c0);
    c1 = wmma16(a, frag_bp(sWp[7], lane, 2, 0, 1), c1);
#pragma unroll
    for (int r = 0; r < 8; ++r) {
      int node = nbase + (kh << 3) + r;
      if (node < N) {
        xw1[node * 32 + n0]      = c0[r];
        xw1[node * 32 + n0 + 16] = c1[r];
      }
    }
  }
}

// ---------------------------------------------------------------------------
// GCN conv helpers
// ---------------------------------------------------------------------------
__global__ void k_fill1(float* __restrict__ d, int n) {
  int i = blockIdx.x * 256 + threadIdx.x;
  if (i < n) d[i] = 1.0f;  // self-loop weight
}

__global__ void k_degacc(float* __restrict__ deg, const int* __restrict__ dst,
                         const float* __restrict__ w, int E) {
  int e = blockIdx.x * 256 + threadIdx.x;
  if (e < E) atomicAdd(&deg[dst[e]], w[e]);
}

__global__ void k_dinv(float* __restrict__ d, int n) {
  int i = blockIdx.x * 256 + threadIdx.x;
  if (i < n) { float v = d[i]; d[i] = v > 0.f ? rsqrtf(v) : 0.f; }
}

__global__ void k_gcn_init(float* __restrict__ h, const float* __restrict__ xw,
                           const float* __restrict__ dinv, const float* __restrict__ bias, int N) {
  int t = blockIdx.x * 256 + threadIdx.x;
  if (t < N * 32) {
    int i = t >> 5, c = t & 31;
    float di = dinv[i];
    h[t] = bias[c] + di * di * xw[t];
  }
}

__global__ void k_gcn_scatter(float* __restrict__ h, const float* __restrict__ xw,
                              const float* __restrict__ dinv, const int* __restrict__ src,
                              const int* __restrict__ dst, const float* __restrict__ w, int E) {
  int t = blockIdx.x * 256 + threadIdx.x;
  if (t < E * 32) {
    int e = t >> 5, c = t & 31;
    int s = src[e], d = dst[e];
    float norm = dinv[s] * w[e] * dinv[d];
    atomicAdd(&h[d * 32 + c], norm * xw[s * 32 + c]);
  }
}

// ---------------------------------------------------------------------------
// Kernel: edgenet MLP  (concat(h[src],h[dst]) -> 64->32->32->1, sigmoid)
// ---------------------------------------------------------------------------
__global__ __launch_bounds__(256) void k_edgenet(P47 P, const float* __restrict__ h,
                                                 float* __restrict__ ew, int E) {
  __shared__ _Float16 sW0p[2048];  // 64x32 packed
  __shared__ _Float16 sW1p[1024];  // 32x32 packed
  __shared__ float sW2[32];
  __shared__ float sB0[32], sB1[32];
  __shared__ float sB2v;
  __shared__ _Float16 sAct[8][16 * 72];

  stage_pack(sW0p, (const float*)P.p[21], 64, 32);
  stage_pack(sW1p, (const float*)P.p[23], 32, 32);
  if (threadIdx.x < 32) {
    sW2[threadIdx.x] = ((const float*)P.p[25])[threadIdx.x];
    sB0[threadIdx.x] = ((const float*)P.p[22])[threadIdx.x];
    sB1[threadIdx.x] = ((const float*)P.p[24])[threadIdx.x];
  }
  if (threadIdx.x == 0) sB2v = ((const float*)P.p[26])[0];
  __syncthreads();

  int wave = threadIdx.x >> 5, lane = threadIdx.x & 31;
  int ntiles = (E + 15) >> 4;
  int tile = blockIdx.x * 8 + wave;
  if (tile >= ntiles) return;

  const int* srcA = (const int*)P.p[1];
  const int* dstA = srcA + E;
  _Float16* act = sAct[wave];
  int ebase = tile << 4;

  // gather 16 x 64 (h[src] | h[dst]) as f16, float4 granularity
  for (int idx = lane; idx < 256; idx += 32) {
    int r = idx >> 4, q = idx & 15;       // q-th float4 of the 64-wide row
    int e = ebase + r; if (e > E - 1) e = E - 1;
    int node = (q < 8) ? srcA[e] : dstA[e];
    float4 val = ((const float4*)(h + node * 32))[q & 7];
    *(v4h*)(act + r * 72 + q * 4) =
        (v4h){(_Float16)val.x, (_Float16)val.y, (_Float16)val.z, (_Float16)val.w};
  }

  int n0 = lane & 15, kh = (lane >> 4) & 1;

  // layer 0: 64 -> 32, SELU  (two K=32 chunks per N-tile)
  {
    v16h a0 = frag_a(act, lane, 72, 0);
    v16h a1 = frag_a(act, lane, 72, 32);
    v8f c0 = {}, c1 = {};
    c0 = wmma16(a0, frag_bp(sW0p, lane, 2, 0, 0), c0);
    c0 = wmma16(a1, frag_bp(sW0p, lane, 2, 1, 0), c0);
    c1 = wmma16(a0, frag_bp(sW0p, lane, 2, 0, 1), c1);
    c1 = wmma16(a1, frag_bp(sW0p, lane, 2, 1, 1), c1);
#pragma unroll
    for (int r = 0; r < 8; ++r) {
      int m = (kh << 3) + r;
      act[m * 72 + n0]      = (_Float16)selu_f(c0[r] + sB0[n0]);
      act[m * 72 + n0 + 16] = (_Float16)selu_f(c1[r] + sB0[n0 + 16]);
    }
  }
  // layer 1: 32 -> 32, SELU
  {
    v16h a = frag_a(act, lane, 72, 0);
    v8f c0 = {}, c1 = {};
    c0 = wmma16(a, frag_bp(sW1p, lane, 2, 0, 0), c0);
    c1 = wmma16(a, frag_bp(sW1p, lane, 2, 0, 1), c1);
#pragma unroll
    for (int r = 0; r < 8; ++r) {
      int m = (kh << 3) + r;
      act[m * 72 + n0]      = (_Float16)selu_f(c0[r] + sB1[n0]);
      act[m * 72 + n0 + 16] = (_Float16)selu_f(c1[r] + sB1[n0 + 16]);
    }
  }
  // layer 2: 32 -> 1, sigmoid (lane-scalar dot)
  if (lane < 16) {
    int e = ebase + lane;
    if (e < E) {
      float acc = sB2v;
#pragma unroll
      for (int k = 0; k < 32; ++k) acc += (float)act[lane * 72 + k] * sW2[k];
      ew[e] = 1.f / (1.f + __expf(-acc));
    }
  }
}

// ---------------------------------------------------------------------------
// Kernel: xw = hin @ W  (32x32, no bias) for conv2
// ---------------------------------------------------------------------------
__global__ __launch_bounds__(256) void k_nodegemm(const float* __restrict__ W,
                                                  const float* __restrict__ hin,
                                                  float* __restrict__ out, int N) {
  __shared__ _Float16 sWp[1024];
  __shared__ _Float16 sAct[8][16 * 40];
  stage_pack(sWp, W, 32, 32);
  __syncthreads();

  int wave = threadIdx.x >> 5, lane = threadIdx.x & 31;
  int ntiles = (N + 15) >> 4;
  int tile = blockIdx.x * 8 + wave;
  if (tile >= ntiles) return;

  _Float16* act = sAct[wave];
  int nbase = tile << 4;
  for (int idx = lane; idx < 128; idx += 32) {
    int r = idx >> 3, q = idx & 7;
    int node = nbase + r; if (node > N - 1) node = N - 1;
    float4 val = ((const float4*)(hin + node * 32))[q];
    *(v4h*)(act + r * 40 + q * 4) =
        (v4h){(_Float16)val.x, (_Float16)val.y, (_Float16)val.z, (_Float16)val.w};
  }

  v16h a = frag_a(act, lane, 40, 0);
  v8f c0 = {}, c1 = {};
  c0 = wmma16(a, frag_bp(sWp, lane, 2, 0, 0), c0);
  c1 = wmma16(a, frag_bp(sWp, lane, 2, 0, 1), c1);
  int n0 = lane & 15, kh = (lane >> 4) & 1;
#pragma unroll
  for (int r = 0; r < 8; ++r) {
    int node = nbase + (kh << 3) + r;
    if (node < N) {
      out[node * 32 + n0]      = c0[r];
      out[node * 32 + n0 + 16] = c1[r];
    }
  }
}

// ---------------------------------------------------------------------------
// Kernel: output MLP  concat(x,h) 35 -> 8x32 (SELU) -> 15 ; split outputs
// ---------------------------------------------------------------------------
__global__ __launch_bounds__(256) void k_out_mlp(P47 P, const float* __restrict__ h,
                                                 float* __restrict__ out_ids,
                                                 float* __restrict__ out_p4, int N) {
  __shared__ float sW0[35 * 32];
  __shared__ float sB0[32];
  __shared__ _Float16 sWp[7][1024];
  __shared__ float sB[7][32];
  __shared__ _Float16 sW8p[512];   // 32x16 packed (col 15 zero-padded)
  __shared__ float sB8[16];
  __shared__ _Float16 sAct[8][16 * 40];
  {
    const float* g = (const float*)P.p[29];
    for (int i = threadIdx.x; i < 1120; i += 256) sW0[i] = g[i];
    if (threadIdx.x < 32) sB0[threadIdx.x] = ((const float*)P.p[30])[threadIdx.x];
    for (int l = 0; l < 7; ++l) {
      stage_pack(sWp[l], (const float*)P.p[31 + 2 * l], 32, 32);
      if (threadIdx.x < 32) sB[l][threadIdx.x] = ((const float*)P.p[32 + 2 * l])[threadIdx.x];
    }
    const float* g8 = (const float*)P.p[45];  // 32x15
    for (int idx = threadIdx.x; idx < 512; idx += 256) {
      int e = idx & 15, n0 = (idx >> 4) & 15, kh = (idx >> 8) & 1;
      int k = kh * 16 + e;
      sW8p[idx] = (n0 < 15) ? (_Float16)g8[k * 15 + n0] : (_Float16)0.f;
    }
    if (threadIdx.x < 16)
      sB8[threadIdx.x] = (threadIdx.x < 15) ? ((const float*)P.p[46])[threadIdx.x] : 0.f;
  }
  __syncthreads();

  int wave = threadIdx.x >> 5, lane = threadIdx.x & 31;
  int ntiles = (N + 15) >> 4;
  int tile = blockIdx.x * 8 + wave;
  if (tile >= ntiles) return;

  const float* x = (const float*)P.p[0];
  _Float16* act = sAct[wave];
  int nbase = tile << 4;

  // layer 0: 35 -> 32, SELU (lane-scalar; input = [x(3) | h(32)])
  for (int idx = lane; idx < 512; idx += 32) {
    int r = idx >> 5, c = idx & 31;
    int node = nbase + r; if (node > N - 1) node = N - 1;
    const float* xr = x + node * 3;
    float acc = sB0[c] + xr[0] * sW0[c] + xr[1] * sW0[32 + c] + xr[2] * sW0[64 + c];
    const float* hr = h + node * 32;
#pragma unroll
    for (int k = 0; k < 32; ++k) acc += hr[k] * sW0[(3 + k) * 32 + c];
    act[r * 40 + c] = (_Float16)selu_f(acc);
  }

  int n0 = lane & 15, kh = (lane >> 4) & 1;

  // layers 1..7: 32 -> 32, SELU on all
  for (int l = 0; l < 7; ++l) {
    v16h a = frag_a(act, lane, 40, 0);
    v8f c0 = {}, c1 = {};
    c0 = wmma16(a, frag_bp(sWp[l], lane, 2, 0, 0), c0);
    c1 = wmma16(a, frag_bp(sWp[l], lane, 2, 0, 1), c1);
#pragma unroll
    for (int r = 0; r < 8; ++r) {
      int m = (kh << 3) + r;
      act[m * 40 + n0]      = (_Float16)selu_f(c0[r] + sB[l][n0]);
      act[m * 40 + n0 + 16] = (_Float16)selu_f(c1[r] + sB[l][n0 + 16]);
    }
  }

  // layer 8: 32 -> 15 (padded to 16), no act; scatter to cand_ids / cand_p4
  {
    v16h a = frag_a(act, lane, 40, 0);
    v8f c0 = {};
    c0 = wmma16(a, frag_bp(sW8p, lane, 1, 0, 0), c0);
#pragma unroll
    for (int r = 0; r < 8; ++r) {
      int node = nbase + (kh << 3) + r;
      if (node < N && n0 < 15) {
        float v = c0[r] + sB8[n0];
        if (n0 < 11) out_ids[node * 11 + n0] = v;
        else         out_p4[node * 4 + (n0 - 11)] = v;
      }
    }
  }
}

// ---------------------------------------------------------------------------
extern "C" void kernel_launch(void* const* d_in, const int* in_sizes, int n_in,
                              void* d_out, int out_size, void* d_ws, size_t ws_size,
                              hipStream_t stream) {
  (void)out_size; (void)ws_size;
  const int N = in_sizes[0] / 3;   // 100000
  const int E = in_sizes[2];       // 1600000 (edge_attr is (E,1))

  P47 P;
  for (int i = 0; i < 47 && i < n_in; ++i) P.p[i] = d_in[i];

  float* xw   = (float*)d_ws;                  // N*32
  float* hbuf = xw + (size_t)N * 32;           // N*32
  float* dinv = hbuf + (size_t)N * 32;         // N

  float* ew      = (float*)d_out;              // [E]
  float* out_ids = ew + E;                     // [N,11]
  float* out_p4  = out_ids + (size_t)N * 11;   // [N,4]

  const int* srcA = (const int*)d_in[1];
  const int* dstA = srcA + E;
  const float* ea = (const float*)d_in[2];

  const int mlpBlocks  = ((N + 15) / 16 + 7) / 8;
  const int edgeBlocks = ((E + 15) / 16 + 7) / 8;
  const int nB   = (N + 255) / 256;
  const int eB   = (E + 255) / 256;
  const int n32B = (N * 32 + 255) / 256;
  const int e32B = (E * 32 + 255) / 256;

  // h0 = inp_mlp(x); xw = h0 @ conv1_W
  k_inp_mlp<<<mlpBlocks, 256, 0, stream>>>(P, xw, N);

  // conv1: deg / dinv / self-loop+bias init / scatter
  k_fill1<<<nB, 256, 0, stream>>>(dinv, N);
  k_degacc<<<eB, 256, 0, stream>>>(dinv, dstA, ea, E);
  k_dinv<<<nB, 256, 0, stream>>>(dinv, N);
  k_gcn_init<<<n32B, 256, 0, stream>>>(hbuf, xw, dinv, (const float*)d_in[20], N);
  k_gcn_scatter<<<e32B, 256, 0, stream>>>(hbuf, xw, dinv, srcA, dstA, ea, E);

  // edge_weight = edgenet(concat(h[src],h[dst]))
  k_edgenet<<<edgeBlocks, 256, 0, stream>>>(P, hbuf, ew, E);

  // conv2: xw = h1 @ conv2_W, then aggregate with learned edge weights
  k_nodegemm<<<mlpBlocks, 256, 0, stream>>>((const float*)d_in[27], hbuf, xw, N);
  k_fill1<<<nB, 256, 0, stream>>>(dinv, N);
  k_degacc<<<eB, 256, 0, stream>>>(dinv, dstA, ew, E);
  k_dinv<<<nB, 256, 0, stream>>>(dinv, N);
  k_gcn_init<<<n32B, 256, 0, stream>>>(hbuf, xw, dinv, (const float*)d_in[28], N);
  k_gcn_scatter<<<e32B, 256, 0, stream>>>(hbuf, xw, dinv, srcA, dstA, ew, E);

  // r = nn1(concat(x, h2)); split into cand_ids / cand_p4
  k_out_mlp<<<mlpBlocks, 256, 0, stream>>>(P, hbuf, out_ids, out_p4, N);
}
